// MultiHeadAttention_89489938580154
// MI455X (gfx1250) — matrix operand order repaired and checked by reading, not compiled, not run on hardware
//
#include <hip/hip_runtime.h>

// ---------------- problem constants ----------------
#define B_    4
#define L_    2048
#define D_    216
#define H_    4
#define HD_   54
#define M_    64
#define S_    2112      // M_ + L_
#define KP_   224       // D_ padded to multiple of 32 (GEMM K)
#define HDP_  64        // hd padded
#define N1_   648       // 3*D
#define N1P_  704       // padded to 64 (11 tiles of 64)
#define WON_  256       // Wo N padded to 64 (4 tiles of 64)
#define ROWS_ 8192      // B_*L_

typedef __attribute__((ext_vector_type(16))) __bf16 v16bf;
typedef __attribute__((ext_vector_type(8)))  __bf16 v8bf;
typedef __attribute__((ext_vector_type(8)))  float  v8f;

__device__ __forceinline__ __bf16 f2bf(float f) {
  unsigned u = __builtin_bit_cast(unsigned, f);
  unsigned r = u + 0x7FFFu + ((u >> 16) & 1u);     // round-to-nearest-even
  unsigned short h = (unsigned short)(r >> 16);
  return __builtin_bit_cast(__bf16, h);
}

__device__ __forceinline__ v8f wmma_bf16(v16bf a, v16bf b, v8f c) {
  return __builtin_amdgcn_wmma_f32_16x16x32_bf16(false, a, false, b, (short)0, c,
                                                 false, false);
}

// ---------------- conversion / prep kernels ----------------
// xb[row][k] (bf16, K padded) and zero the padded attn-output buffer (same shape)
__global__ void k_conv_x(const float* __restrict__ x, __bf16* __restrict__ xb,
                         __bf16* __restrict__ ab2) {
  int i = blockIdx.x * blockDim.x + threadIdx.x;
  if (i >= ROWS_ * KP_) return;
  int row = i / KP_, k = i % KP_;
  xb[i]  = (k < D_) ? f2bf(x[(size_t)row * D_ + k]) : f2bf(0.f);
  ab2[i] = f2bf(0.f);
}

// WqkvT[n][k] = Wqkv[k][n] (N-major so B-fragments load contiguously)
__global__ void k_conv_wqkv(const float* __restrict__ w, __bf16* __restrict__ wt) {
  int i = blockIdx.x * blockDim.x + threadIdx.x;
  if (i >= N1P_ * KP_) return;
  int n = i / KP_, k = i % KP_;
  float v = (n < N1_ && k < D_) ? w[(size_t)k * N1_ + n] : 0.f;
  wt[i] = f2bf(v);
}

__global__ void k_conv_wo(const float* __restrict__ w, __bf16* __restrict__ wt) {
  int i = blockIdx.x * blockDim.x + threadIdx.x;
  if (i >= WON_ * KP_) return;
  int n = i / KP_, k = i % KP_;
  float v = (n < D_ && k < D_) ? w[(size_t)k * D_ + n] : 0.f;
  wt[i] = f2bf(v);
}

// additive key-padding bias per (b, kv position)
__global__ void k_maskf(const int* __restrict__ am, float* __restrict__ mf) {
  int i = blockIdx.x * blockDim.x + threadIdx.x;
  if (i >= B_ * S_) return;
  int b = i / S_, s = i % S_;
  mf[i] = (s < M_) ? 0.f : (am[b * L_ + (s - M_)] ? 0.f : -1e30f);
}

// ---------------- bf16 WMMA GEMM: one wave = 16(M) x 64(N) tile ----------------
// C[M x realN] f32 = A[M x Kp](row-major bf16) * Bt[Np x Kp](N-major bf16)
// A-fragment reused across 4 N-subtiles -> 4 WMMAs per (32B A + 128B B) loaded.
__global__ __launch_bounds__(256) void k_gemm(const __bf16* __restrict__ A,
                                              const __bf16* __restrict__ Bt,
                                              float* __restrict__ C,
                                              int Mtiles, int N64tiles, int Kp,
                                              int ldc, int realN) {
  int wave = (blockIdx.x * blockDim.x + threadIdx.x) >> 5;
  int lane = threadIdx.x & 31;
  int nt = wave % N64tiles, mt = wave / N64tiles;
  if (mt >= Mtiles) return;
  int sub = lane & 15, half = lane >> 4;
  const __bf16* ap = A + (size_t)(mt * 16 + sub) * Kp + half * 16;
  const __bf16* bp = Bt + (size_t)(nt * 64 + sub) * Kp + half * 16;
  v8f acc[4] = {};
  for (int k0 = 0; k0 < Kp; k0 += 32) {
    v8bf alo = *(const v8bf*)(ap + k0);
    v8bf ahi = *(const v8bf*)(ap + k0 + 8);
    v16bf a;
#pragma unroll
    for (int i = 0; i < 8; ++i) { a[i] = alo[i]; a[8 + i] = ahi[i]; }
    v16bf b0 = *(const v16bf*)(bp + k0);
    v16bf b1 = *(const v16bf*)(bp + (size_t)16 * Kp + k0);
    v16bf b2 = *(const v16bf*)(bp + (size_t)32 * Kp + k0);
    v16bf b3 = *(const v16bf*)(bp + (size_t)48 * Kp + k0);
    acc[0] = wmma_bf16(a, b0, acc[0]);
    acc[1] = wmma_bf16(a, b1, acc[1]);
    acc[2] = wmma_bf16(a, b2, acc[2]);
    acc[3] = wmma_bf16(a, b3, acc[3]);
  }
#pragma unroll
  for (int t = 0; t < 4; ++t) {
    int n = nt * 64 + t * 16 + sub;
    if (n < realN) {
#pragma unroll
      for (int r = 0; r < 8; ++r)
        C[(size_t)(mt * 16 + half * 8 + r) * ldc + n] = acc[t][r];
    }
  }
}

// ---------------- RoPE + KV assembly ----------------
// qb[bh][l][64] = RoPE(q)*scale ; kall[bh][s][64] = {mem_k, RoPE(k)} ;
// vT[bh][d][s]  = {mem_v, v}  (d-major so PV B-fragments are contiguous)
__global__ void k_assemble(const float* __restrict__ qkv,
                           const float* __restrict__ mem_k,
                           const float* __restrict__ mem_v,
                           __bf16* __restrict__ qb, __bf16* __restrict__ kall,
                           __bf16* __restrict__ vT) {
  long long i = (long long)blockIdx.x * blockDim.x + threadIdx.x;
  if (i >= (long long)B_ * H_ * S_ * HDP_) return;
  int d  = (int)(i & 63);
  int s  = (int)((i >> 6) % S_);
  int bh = (int)(i / ((long long)S_ * HDP_));
  int h = bh % H_, b = bh / H_;

  float kv = 0.f, vv = 0.f;
  if (d < HD_) {
    if (s < M_) {
      size_t mi = ((size_t)b * M_ + s) * D_ + h * HD_ + d;
      kv = mem_k[mi];
      vv = mem_v[mi];
    } else {
      int l = s - M_;
      size_t base = ((size_t)(b * L_ + l) * 3) * D_ + h * HD_;
      int dm = d % 27;
      float invf = __powf(10000.f, -(2.f * dm) / 54.f);
      float c, sn; __sincosf((float)l * invf, &sn, &c);
      // k with RoPE
      float kd = qkv[base + D_ + d];
      float ko = (d < 27) ? -qkv[base + D_ + d + 27] : qkv[base + D_ + d - 27];
      kv = kd * c + ko * sn;
      vv = qkv[base + 2 * D_ + d];
      // q with RoPE, softmax scale folded in
      float qd = qkv[base + d];
      float qo = (d < 27) ? -qkv[base + d + 27] : qkv[base + d - 27];
      float qr = (qd * c + qo * sn) * 0.13608276348795434f; // 54^-0.5
      qb[((size_t)bh * L_ + l) * HDP_ + d] = f2bf(qr);
    }
  } else if (s >= M_) {
    qb[((size_t)bh * L_ + (s - M_)) * HDP_ + d] = f2bf(0.f);
  }
  kall[((size_t)bh * S_ + s) * HDP_ + d] = f2bf(kv);
  vT[((size_t)bh * HDP_ + d) * S_ + s]   = f2bf(vv);
}

// ---------------- flash attention (one wave = one 16-row Q tile) ----------------
__global__ __launch_bounds__(256) void k_flash(const __bf16* __restrict__ qb,
                                               const __bf16* __restrict__ kall,
                                               const __bf16* __restrict__ vT,
                                               const float* __restrict__ maskf,
                                               __bf16* __restrict__ ab2) {
  __shared__ __bf16 pshare[8][16 * 32];   // per-wave P transpose slab
  int lane = threadIdx.x & 31;
  int wv   = threadIdx.x >> 5;
  int gw   = blockIdx.x * 8 + wv;         // 2048 waves total
  int qt = gw & 127, bh = gw >> 7;
  int b = bh >> 2, h = bh & 3;
  int sub = lane & 15, half = lane >> 4;
  int l_base = qt * 16;

  // Q fragments (rows l = l_base+sub), K-chunks d[0..31], d[32..63]
  const __bf16* qp = qb + ((size_t)bh * L_ + l_base + sub) * HDP_;
  v16bf qa0, qa1;
  {
    v8bf x0 = *(const v8bf*)(qp + half * 16);
    v8bf x1 = *(const v8bf*)(qp + 8 + half * 16);
    v8bf x2 = *(const v8bf*)(qp + 32 + half * 16);
    v8bf x3 = *(const v8bf*)(qp + 40 + half * 16);
#pragma unroll
    for (int i = 0; i < 8; ++i) {
      qa0[i] = x0[i]; qa0[8 + i] = x1[i];
      qa1[i] = x2[i]; qa1[8 + i] = x3[i];
    }
  }

  float mrow[8], lrow[8], alpha[8];
#pragma unroll
  for (int r = 0; r < 8; ++r) { mrow[r] = -1e30f; lrow[r] = 0.f; }
  v8f o0 = {}, o1 = {}, o2 = {}, o3 = {};

  const __bf16* kbase = kall + (size_t)bh * S_ * HDP_;
  const __bf16* vbase = vT + (size_t)bh * HDP_ * S_;
  const int s_end = M_ + l_base + 16;     // causal truncation (exclusive)

  for (int s0 = 0; s0 < s_end; s0 += 32) {
    v8f sc[2];
#pragma unroll
    for (int t = 0; t < 2; ++t) {
      int scol = s0 + t * 16 + sub;
      const __bf16* kp = kbase + (size_t)scol * HDP_ + half * 16;
      v16bf kb0 = *(const v16bf*)(kp);
      v16bf kb1 = *(const v16bf*)(kp + 32);
      v8f c = {};
      c = wmma_bf16(qa0, kb0, c);
      c = wmma_bf16(qa1, kb1, c);
      float mk = maskf[b * S_ + scol];
#pragma unroll
      for (int r = 0; r < 8; ++r) {
        int lg = l_base + half * 8 + r;
        float v = c[r] + mk;
        if (scol >= M_ + lg + 1) v = -1e30f;  // causal
        c[r] = v;
      }
      sc[t] = c;
    }
    // online softmax: per-row max / sum across the 16-lane column group
#pragma unroll
    for (int r = 0; r < 8; ++r) {
      float mx = fmaxf(sc[0][r], sc[1][r]);
      mx = fmaxf(mx, __shfl_xor(mx, 8, 32));
      mx = fmaxf(mx, __shfl_xor(mx, 4, 32));
      mx = fmaxf(mx, __shfl_xor(mx, 2, 32));
      mx = fmaxf(mx, __shfl_xor(mx, 1, 32));
      float mnew = fmaxf(mrow[r], mx);
      float a = __expf(mrow[r] - mnew);
      float p0 = __expf(sc[0][r] - mnew), p1 = __expf(sc[1][r] - mnew);
      sc[0][r] = p0; sc[1][r] = p1;
      float rs = p0 + p1;
      rs += __shfl_xor(rs, 8, 32);
      rs += __shfl_xor(rs, 4, 32);
      rs += __shfl_xor(rs, 2, 32);
      rs += __shfl_xor(rs, 1, 32);
      lrow[r] = lrow[r] * a + rs;
      mrow[r] = mnew;
      alpha[r] = a;
    }
    // transpose P (C-layout -> A-layout) through per-wave LDS, as bf16
    __bf16* pl = &pshare[wv][0];
#pragma unroll
    for (int r = 0; r < 8; ++r) {
      int m = half * 8 + r;
      pl[m * 32 + sub]      = f2bf(sc[0][r]);
      pl[m * 32 + 16 + sub] = f2bf(sc[1][r]);
    }
    asm volatile("s_wait_dscnt 0" ::: "memory");
    v16bf pa;
    {
      v8bf lo = *(const v8bf*)(pl + sub * 32 + half * 16);
      v8bf hi = *(const v8bf*)(pl + sub * 32 + 8 + half * 16);
#pragma unroll
      for (int i = 0; i < 8; ++i) { pa[i] = lo[i]; pa[8 + i] = hi[i]; }
    }
    // rescale accumulators, then O += P @ V (4 d-tiles of 16)
#pragma unroll
    for (int r = 0; r < 8; ++r) {
      o0[r] *= alpha[r]; o1[r] *= alpha[r];
      o2[r] *= alpha[r]; o3[r] *= alpha[r];
    }
    const __bf16* vp = vbase + s0 + half * 16;
    o0 = wmma_bf16(pa, *(const v16bf*)(vp + (size_t)(sub)      * S_), o0);
    o1 = wmma_bf16(pa, *(const v16bf*)(vp + (size_t)(16 + sub) * S_), o1);
    o2 = wmma_bf16(pa, *(const v16bf*)(vp + (size_t)(32 + sub) * S_), o2);
    o3 = wmma_bf16(pa, *(const v16bf*)(vp + (size_t)(48 + sub) * S_), o3);
  }

  // normalize, write padded bf16 attn output [row=b*L+l][col=h*54+d]
  const v8f* ot[4] = { &o0, &o1, &o2, &o3 };
#pragma unroll
  for (int t = 0; t < 4; ++t) {
    int d = t * 16 + sub;
    if (d < HD_) {
#pragma unroll
      for (int r = 0; r < 8; ++r) {
        int lg = l_base + half * 8 + r;
        float v = (*ot[t])[r] / lrow[r];
        ab2[((size_t)(b * L_ + lg)) * KP_ + h * HD_ + d] = f2bf(v);
      }
    }
  }
}

// ---------------- launch ----------------
extern "C" void kernel_launch(void* const* d_in, const int* in_sizes, int n_in,
                              void* d_out, int out_size, void* d_ws, size_t ws_size,
                              hipStream_t stream) {
  (void)in_sizes; (void)n_in; (void)out_size; (void)ws_size;
  const float* x     = (const float*)d_in[0];
  const float* mem_k = (const float*)d_in[1];
  const float* mem_v = (const float*)d_in[2];
  const int*   amask = (const int*)d_in[3];
  const float* Wqkv  = (const float*)d_in[4];
  const float* Wo    = (const float*)d_in[5];
  float* out = (float*)d_out;

  // workspace layout (bytes, 256-aligned)
  char* ws = (char*)d_ws;
  size_t off = 0;
  auto take = [&](size_t bytes) { char* p = ws + off; off = (off + bytes + 255) & ~(size_t)255; return p; };
  __bf16* xb     = (__bf16*)take((size_t)ROWS_ * KP_ * 2);
  __bf16* wqkvT  = (__bf16*)take((size_t)N1P_ * KP_ * 2);
  __bf16* woT    = (__bf16*)take((size_t)WON_ * KP_ * 2);
  float*  qkv    = (float*)take((size_t)ROWS_ * N1_ * 4);
  __bf16* qb     = (__bf16*)take((size_t)B_ * H_ * L_ * HDP_ * 2);
  __bf16* kall   = (__bf16*)take((size_t)B_ * H_ * S_ * HDP_ * 2);
  __bf16* vT     = (__bf16*)take((size_t)B_ * H_ * HDP_ * S_ * 2);
  float*  maskf  = (float*)take((size_t)B_ * S_ * 4);
  __bf16* ab2    = (__bf16*)take((size_t)ROWS_ * KP_ * 2);

  // 1. conversions / padding / mask bias
  {
    int n = ROWS_ * KP_;
    k_conv_x<<<(n + 255) / 256, 256, 0, stream>>>(x, xb, ab2);
  }
  { int n = N1P_ * KP_; k_conv_wqkv<<<(n + 255) / 256, 256, 0, stream>>>(Wqkv, wqkvT); }
  { int n = WON_ * KP_; k_conv_wo<<<(n + 255) / 256, 256, 0, stream>>>(Wo, woT); }
  { int n = B_ * S_;    k_maskf<<<(n + 255) / 256, 256, 0, stream>>>(amask, maskf); }

  // 2. GEMM1: qkv[8192 x 648] = xb @ WqkvT    (512 M-tiles x 11 N64-tiles)
  {
    int waves = 512 * (N1P_ / 64);
    int blocks = (waves * 32 + 255) / 256;
    k_gemm<<<blocks, 256, 0, stream>>>(xb, wqkvT, qkv, 512, N1P_ / 64, KP_, N1_, N1_);
  }

  // 3. RoPE + KV assembly
  {
    long long n = (long long)B_ * H_ * S_ * HDP_;
    k_assemble<<<(int)((n + 255) / 256), 256, 0, stream>>>(qkv, mem_k, mem_v, qb, kall, vT);
  }

  // 4. flash attention: 2048 waves, 8 per block
  k_flash<<<256, 256, 0, stream>>>(qb, kall, vT, maskf, ab2);

  // 5. GEMM2: out[8192 x 216] = ab2 @ WoT    (512 M-tiles x 4 N64-tiles)
  {
    int waves = 512 * (WON_ / 64);
    int blocks = (waves * 32 + 255) / 256;
    k_gemm<<<blocks, 256, 0, stream>>>(ab2, woT, out, 512, WON_ / 64, KP_, D_, D_);
  }
}
